// MultiDConvHeadAttention_4853313045251
// MI455X (gfx1250) — compile-verified, matrix operand order checked by reading, not run on hardware
//
#include <hip/hip_runtime.h>
#include <hip/hip_bf16.h>
#include <math.h>

// ---------------------------------------------------------------------------
// MultiDConvHeadAttention for MI455X (gfx1250, wave32, WMMA bf16 16x16x32)
// Operands for all matmuls pre-packed into WMMA-fragment-major bf16 so each
// lane loads whole fragments with contiguous B128 loads. Working set ~30 MB
// => L2-resident (192 MB); hot loops are WMMA-issue-bound.
// GEMM waves carry TWO independent accumulator chains (2 N-tiles sharing the
// A fragment) to fill the XDL pipeline and halve A traffic.
// ---------------------------------------------------------------------------

typedef __attribute__((ext_vector_type(16))) __bf16 v16bf;
typedef __attribute__((ext_vector_type(8)))  __bf16 v8bf;
typedef __attribute__((ext_vector_type(8)))  float  v8f;

#define HW   4096
#define CDIM 192
#define QKVC 576
#define NHEAD 6
#define HDIM 32
#define ATTN_SCALE 0.17677669529663687f  /* 32^-0.5 */

// Fragment element mapping (ISA 7.12.2, 16-bit A/B 16x32):
//   lane = (r, hi):  element e<8  -> k = hi*8 + e
//                    element e>=8 -> k = 16 + hi*8 + (e-8)
__device__ __forceinline__ int frag_k(int hi, int e) {
    return (e < 8) ? (hi * 8 + e) : (16 + hi * 8 + (e - 8));
}

__device__ __forceinline__ v16bf cat8(v8bf lo, v8bf hi) {
    return __builtin_shufflevector(lo, hi, 0, 1, 2, 3, 4, 5, 6, 7,
                                   8, 9, 10, 11, 12, 13, 14, 15);
}

// ===========================================================================
// Weight f32 -> bf16 row-major (qkv_w 576x192, po_w 192x192).
// ===========================================================================
__global__ void convert_weights(const float* __restrict__ w1,
                                const float* __restrict__ w2,
                                __bf16* __restrict__ o1,
                                __bf16* __restrict__ o2)
{
    const int idx = blockIdx.x * blockDim.x + threadIdx.x;  // < 147456
    if (idx < QKVC * CDIM) o1[idx] = (__bf16)w1[idx];
    else                   o2[idx - QKVC * CDIM] = (__bf16)w2[idx - QKVC * CDIM];
}

// ===========================================================================
// Pack a (K x 4096) f32 matrix into B-fragment-major bf16.
// Fragment id = nt*(K/32) + kt; per fragment: 32 lanes x 16 bf16 contiguous.
// Gather-read, coalesced-write; done once per matrix.
// ===========================================================================
__global__ void pack_bfrag(const float* __restrict__ X,
                           __bf16* __restrict__ dst, int K)
{
    const int idx = blockIdx.x * blockDim.x + threadIdx.x;  // < K*4096
    const int e    = idx & 15;
    const int lane = (idx >> 4) & 31;
    const int frag = idx >> 9;
    const int ktiles = K >> 5;
    const int nt = frag / ktiles;
    const int kt = frag % ktiles;
    const int r = lane & 15, hi = lane >> 4;
    const int k = kt * 32 + frag_k(hi, e);
    const int n = nt * 16 + r;
    dst[idx] = (__bf16)X[(size_t)k * HW + n];
}

// ===========================================================================
// GEMM: C(MxN) = W(MxK) * X(KxN) + bias.  W: bf16 row-major (A-frag = two
// aligned 16B loads).  X: fragment-packed bf16 (B-frag = one 32B load).
// One wave computes a 16x32 strip: two N-tiles sharing A, two independent
// WMMA accumulator chains.
// ===========================================================================
__global__ void gemm_wmma_bf16(const __bf16* __restrict__ Wb,
                               const __bf16* __restrict__ Xfrag,
                               const float* __restrict__ bias,
                               float* __restrict__ C,
                               int M, int N, int K)
{
    const int lane = threadIdx.x & 31;
    const int wv   = threadIdx.x >> 5;
    const int wid  = blockIdx.x * (blockDim.x >> 5) + wv;
    const int npairs = N >> 5;                 // pairs of 16-col tiles
    const int mt  = wid / npairs;
    const int ntp = wid % npairs;
    if (mt * 16 >= M) return;
    const int r = lane & 15, hi = lane >> 4;
    const int m0 = mt * 16, n0 = ntp * 32;
    const int ktiles = K >> 5;

    const __bf16* wrow = Wb + (size_t)(m0 + r) * K;
    const v16bf*  bp0  = (const v16bf*)Xfrag
                       + (size_t)(2 * ntp) * ktiles * 32 + lane;
    const v16bf*  bp1  = bp0 + (size_t)ktiles * 32;

    v8f acc0 = {}, acc1 = {};
    for (int kt = 0; kt < ktiles; ++kt) {
        const v16bf a = cat8(*(const v8bf*)(wrow + hi * 8),
                             *(const v8bf*)(wrow + 16 + hi * 8));
        acc0 = __builtin_amdgcn_wmma_f32_16x16x32_bf16(
            false, a, false, *bp0, (short)0, acc0, false, false);
        acc1 = __builtin_amdgcn_wmma_f32_16x16x32_bf16(
            false, a, false, *bp1, (short)0, acc1, false, false);
        wrow += 32;
        bp0  += 32;
        bp1  += 32;
    }
#pragma unroll
    for (int v = 0; v < 8; ++v) {
        const int m = m0 + v + 8 * hi;
        const float bi = bias[m];
        C[(size_t)m * N + n0 + r]      = acc0[v] + bi;
        C[(size_t)m * N + n0 + 16 + r] = acc1[v] + bi;
    }
}

// ===========================================================================
// Depthwise 3x3, padding 1, 576 channels of 64x64.
// ===========================================================================
__global__ void dwconv3x3(const float* __restrict__ in,
                          const float* __restrict__ w,
                          const float* __restrict__ b,
                          float* __restrict__ out)
{
    const int idx = blockIdx.x * blockDim.x + threadIdx.x;  // < 576*4096
    const int o = idx >> 12;
    const int n = idx & 4095;
    const int y = n >> 6, x = n & 63;
    const float* src = in + (size_t)o * HW;
    const float* wr  = w + o * 9;
    float acc = b[o];
#pragma unroll
    for (int dy = -1; dy <= 1; ++dy) {
        const int yy = y + dy;
        if ((unsigned)yy >= 64u) continue;
#pragma unroll
        for (int dx = -1; dx <= 1; ++dx) {
            const int xx = x + dx;
            if ((unsigned)xx >= 64u) continue;
            acc += src[yy * 64 + xx] * wr[(dy + 1) * 3 + (dx + 1)];
        }
    }
    out[idx] = acc;
}

// ===========================================================================
// Per-channel L2 norm over spatial axis for q,k channels (384 rows).
// ===========================================================================
__global__ void l2norms(const float* __restrict__ qkvd, float* __restrict__ norms)
{
    const int c = blockIdx.x;
    const float* row = qkvd + (size_t)c * HW;
    float s = 0.f;
    for (int i = threadIdx.x; i < HW; i += blockDim.x) {
        const float v = row[i];
        s += v * v;
    }
    __shared__ float red[256];
    red[threadIdx.x] = s;
    __syncthreads();
    for (int off = blockDim.x >> 1; off > 0; off >>= 1) {
        if ((int)threadIdx.x < off) red[threadIdx.x] += red[threadIdx.x + off];
        __syncthreads();
    }
    if (threadIdx.x == 0) norms[c] = sqrtf(red[0]);
}

// ===========================================================================
// Normalize + scale + pack q/k/v into fragment-major bf16.
//   q,k frag id = h*256 + t        (rows = tokens t*16+r, K = 32 channels)
//   v   frag id = h*256 + u*2 + cg (rows = channels cg*16+r, K = 32 tokens)
// ===========================================================================
__global__ void pack_qkv_frags(const float* __restrict__ qkvd,
                               const float* __restrict__ norms,
                               __bf16* __restrict__ qf,
                               __bf16* __restrict__ kf,
                               __bf16* __restrict__ vf)
{
    const int idx = blockIdx.x * blockDim.x + threadIdx.x;  // < 3*786432
    const int region = idx / (CDIM * HW);
    const int j = idx - region * (CDIM * HW);
    const int e    = j & 15;
    const int lane = (j >> 4) & 31;
    const int frag = j >> 9;
    const int h   = frag >> 8;
    const int rem = frag & 255;
    const int r = lane & 15, hi = lane >> 4;

    if (region < 2) {                       // q or k: rows = tokens
        const int c = h * HDIM + frag_k(hi, e);
        const int n = rem * 16 + r;
        const int ch = region * CDIM + c;
        const float nm = fmaxf(norms[ch], 1e-12f);
        const float v = qkvd[(size_t)ch * HW + n] / nm;
        if (region == 0) qf[j] = (__bf16)(v * ATTN_SCALE);
        else             kf[j] = (__bf16)v;
    } else {                                // v: rows = channels
        const int u = rem >> 1, cg = rem & 1;
        const int c = h * HDIM + cg * 16 + r;
        const int m = u * 32 + frag_k(hi, e);
        vf[j] = (__bf16)qkvd[(size_t)(2 * CDIM + c) * HW + m];
    }
}

// ===========================================================================
// Fused flash attention. Block = 4 waves = one head x 64 query tokens; each
// wave owns 16 query rows, kv swept in 64-token tiles with online softmax.
// All fragments are single 32B loads; S-tiles and the o0/o1 accumulators are
// independent WMMA chains. P turns C/D->A layout through a per-wave LDS tile
// (scalar b16 stores, vectorized ds_load_b128 reads, explicit s_wait_dscnt).
// ===========================================================================
__global__ void flash_attn(const __bf16* __restrict__ Qf,
                           const __bf16* __restrict__ Kf,
                           const __bf16* __restrict__ Vf,
                           float* __restrict__ O)
{
    const int lane = threadIdx.x & 31;
    const int wv   = threadIdx.x >> 5;
    const int r = lane & 15, hi = lane >> 4;
    const int head  = blockIdx.x >> 6;
    const int qtile = (blockIdx.x & 63) * 4 + wv;   // 16-token tile index
    const int q0    = qtile * 16;

    const v16bf* qf = (const v16bf*)Qf + (size_t)head * 256 * 32;
    const v16bf* kp = (const v16bf*)Kf + (size_t)head * 256 * 32 + lane;
    const v16bf* vp = (const v16bf*)Vf + (size_t)head * 256 * 32 + lane;

    __shared__ alignas(16) __bf16 Pld[4][16][80];   // 16B-aligned rows (160B)

    const v16bf aq = qf[(size_t)qtile * 32 + lane];

    v8f o0 = {}, o1 = {};
    float mrow[8], lrow[8];
#pragma unroll
    for (int v = 0; v < 8; ++v) { mrow[v] = -1e30f; lrow[v] = 0.f; }

    for (int mt = 0; mt < 64; ++mt) {               // 64-token kv tiles
        // prefetch next tile's K fragments (global_prefetch_b8)
        __builtin_prefetch(kp + (size_t)(mt + 1) * 128, 0, 1);

        // ---- S = Q^T K : 4 independent wmma ----
        v8f s[4];
#pragma unroll
        for (int g = 0; g < 4; ++g) {
            const v16bf bk = kp[(size_t)(mt * 4 + g) * 32];
            v8f z = {};
            s[g] = __builtin_amdgcn_wmma_f32_16x16x32_bf16(
                false, aq, false, bk, (short)0, z, false, false);
        }

        // ---- online softmax (rows split across 16-lane halves) ----
#pragma unroll
        for (int v = 0; v < 8; ++v) {
            float tmax = fmaxf(fmaxf(s[0][v], s[1][v]), fmaxf(s[2][v], s[3][v]));
#pragma unroll
            for (int m = 1; m < 16; m <<= 1)
                tmax = fmaxf(tmax, __shfl_xor(tmax, m, 32));
            const float nmax  = fmaxf(mrow[v], tmax);
            const float alpha = __expf(mrow[v] - nmax);
            float tsum = 0.f;
#pragma unroll
            for (int g = 0; g < 4; ++g) {
                const float p = __expf(s[g][v] - nmax);
                s[g][v] = p;
                tsum += p;
            }
#pragma unroll
            for (int m = 1; m < 16; m <<= 1)
                tsum += __shfl_xor(tsum, m, 32);
            lrow[v] = lrow[v] * alpha + tsum;
            mrow[v] = nmax;
            o0[v] *= alpha;
            o1[v] *= alpha;
        }

        // ---- P: C/D layout -> LDS -> A layout ----
#pragma unroll
        for (int g = 0; g < 4; ++g)
#pragma unroll
            for (int v = 0; v < 8; ++v)
                Pld[wv][v + 8 * hi][g * 16 + r] = (__bf16)s[g][v];
        asm volatile("s_wait_dscnt 0x0" ::: "memory");

        const __bf16* prow = &Pld[wv][r][0];
        const v16bf ap0 = cat8(*(const v8bf*)(prow + hi * 8),
                               *(const v8bf*)(prow + 16 + hi * 8));
        const v16bf ap1 = cat8(*(const v8bf*)(prow + 32 + hi * 8),
                               *(const v8bf*)(prow + 48 + hi * 8));

        // ---- O += P * V^T : two independent 2-wmma chains ----
        const size_t fb = (size_t)mt * 4;           // v-frag index base
        const v16bf bv00 = vp[(fb + 0) * 32];       // chunk u0,   cg 0
        const v16bf bv01 = vp[(fb + 1) * 32];       // chunk u0,   cg 1
        const v16bf bv10 = vp[(fb + 2) * 32];       // chunk u0+1, cg 0
        const v16bf bv11 = vp[(fb + 3) * 32];       // chunk u0+1, cg 1
        o0 = __builtin_amdgcn_wmma_f32_16x16x32_bf16(
            false, ap0, false, bv00, (short)0, o0, false, false);
        o1 = __builtin_amdgcn_wmma_f32_16x16x32_bf16(
            false, ap0, false, bv01, (short)0, o1, false, false);
        o0 = __builtin_amdgcn_wmma_f32_16x16x32_bf16(
            false, ap1, false, bv10, (short)0, o0, false, false);
        o1 = __builtin_amdgcn_wmma_f32_16x16x32_bf16(
            false, ap1, false, bv11, (short)0, o1, false, false);
    }

    // ---- finalize: O /= l, store channel-major (c, n) f32 ----
#pragma unroll
    for (int v = 0; v < 8; ++v) {
        const float inv = 1.0f / lrow[v];
        const int n = q0 + v + 8 * hi;
        O[(size_t)(head * HDIM + r) * HW + n]      = o0[v] * inv;
        O[(size_t)(head * HDIM + 16 + r) * HW + n] = o1[v] * inv;
    }
}

// ===========================================================================
// Launch
// ===========================================================================
extern "C" void kernel_launch(void* const* d_in, const int* in_sizes, int n_in,
                              void* d_out, int out_size, void* d_ws, size_t ws_size,
                              hipStream_t stream)
{
    const float* x     = (const float*)d_in[0];   // (192, 4096)
    const float* qkv_w = (const float*)d_in[1];   // (576, 192)
    const float* qkv_b = (const float*)d_in[2];   // (576,)
    const float* dw_w  = (const float*)d_in[3];   // (576, 9)
    const float* dw_b  = (const float*)d_in[4];   // (576,)
    const float* po_w  = (const float*)d_in[5];   // (192, 192)
    const float* po_b  = (const float*)d_in[6];   // (192,)
    float* out = (float*)d_out;                   // (192, 4096)

    char* ws = (char*)d_ws;
    float*  qkv   = (float*)(ws);                 // 576*4096 f32
    float*  qkvd  = (float*)(ws + 9437184);       // 576*4096 f32
    float*  norms = (float*)(ws + 18874368);      // 384 f32 (padded)
    __bf16* qfrag = (__bf16*)(ws + 18878464);     // 192*4096 bf16 frag-major
    __bf16* kfrag = (__bf16*)(ws + 20451328);     // 192*4096 bf16 frag-major
    __bf16* vfrag = (__bf16*)(ws + 22024192);     // 192*4096 bf16 frag-major
    float*  aout  = (float*)(ws + 23597056);      // 192*4096 f32
    __bf16* xfrag = (__bf16*)(ws + 26742784);     // 192*4096 bf16 frag-major
    __bf16* afrag = (__bf16*)(ws + 28315648);     // 192*4096 bf16 frag-major
    __bf16* wqkv  = (__bf16*)(ws + 29888512);     // 576*192 bf16
    __bf16* wpo   = (__bf16*)(ws + 30109696);     // 192*192 bf16

    // 0) weight conversion + x fragment packing
    convert_weights<<<576, 256, 0, stream>>>(qkv_w, po_w, wqkv, wpo);
    pack_bfrag<<<(CDIM * HW) / 256, 256, 0, stream>>>(x, xfrag, CDIM);
    // 1) qkv pointwise GEMM: (576x192)*(192x4096); 36*128 wave-strips
    gemm_wmma_bf16<<<576, 256, 0, stream>>>(wqkv, xfrag, qkv_b, qkv,
                                            QKVC, HW, CDIM);
    // 2) depthwise 3x3
    dwconv3x3<<<(QKVC * HW) / 256, 256, 0, stream>>>(qkv, dw_w, dw_b, qkvd);
    // 3) L2 norms for q,k channels
    l2norms<<<2 * CDIM, 256, 0, stream>>>(qkvd, norms);
    // 4) normalize + fragment-pack q/k/v
    pack_qkv_frags<<<(3 * CDIM * HW) / 256, 256, 0, stream>>>(
        qkvd, norms, qfrag, kfrag, vfrag);
    // 5) fused flash attention: 6 heads x 64 query tiles, 4 waves each
    flash_attn<<<NHEAD * 64, 128, 0, stream>>>(qfrag, kfrag, vfrag, aout);
    // 6) output projection: (192x192)*(192x4096); 12*128 wave-strips
    pack_bfrag<<<(CDIM * HW) / 256, 256, 0, stream>>>(aout, afrag, CDIM);
    gemm_wmma_bf16<<<192, 256, 0, stream>>>(wpo, afrag, po_b, out,
                                            CDIM, HW, CDIM);
}
